// _BaseMultiHeadAttention_10651518894093
// MI455X (gfx1250) — matrix operand order repaired and checked by reading, not compiled, not run on hardware
//
#include <hip/hip_runtime.h>

typedef _Float16 v16h __attribute__((ext_vector_type(16)));
typedef _Float16 v8h  __attribute__((ext_vector_type(8)));
typedef float    v8f  __attribute__((ext_vector_type(8)));
typedef unsigned int v4u __attribute__((ext_vector_type(4)));
typedef int      v8i  __attribute__((ext_vector_type(8)));
typedef int      v4i  __attribute__((ext_vector_type(4)));

#define BB  2
#define SS  2048
#define HH  16
#define DH  64
#define DM  (HH * DH)          // 1024
#define NEG_INF_F (-1e9f)
#define KPAD 72                // 32x64 K tile rows padded to 72 halves (144B, 16B-aligned)
#define VPAD 40                // 64x32 tile rows padded to 40 halves (80B, 16B-aligned)

__device__ __forceinline__ v8f wmma_f16(v16h a, v16h b, v8f c) {
    // D = A(16x32 f16) * B(32x16 f16) + C(16x16 f32)
    return __builtin_amdgcn_wmma_f32_16x16x32_f16(
        false, a, false, b, (short)0, c, false, false);
}

__device__ __forceinline__ size_t qkv_idx(int b, int s, int h, int d) {
    return (((size_t)b * SS + (size_t)s) * HH + (size_t)h) * DH + (size_t)d;
}

__device__ __forceinline__ v16h ld_frag_lds(const _Float16* p) {
    v8h lo = *(const v8h*)p;          // ds_load_b128
    v8h hh = *(const v8h*)(p + 8);    // ds_load_b128
    v16h r;
#pragma unroll
    for (int i = 0; i < 8; ++i) { r[i] = lo[i]; r[8 + i] = hh[i]; }
    return r;
}

// ---------------------------------------------------------------------------
// Kernel 1: causal flash attention.
// Block = 128 threads (4 waves) sharing one (b,h); waves own consecutive
// 16-row Q tiles. K tile and V^T tile staged as f16 in double-buffered LDS:
// global loads for tile t+1 issue into registers while tile t computes
// (loadcnt outstanding across the WMMA body); cvt+LDS-store happen at the top
// of the next iteration. Row-sums computed by an extra P x ones WMMA.
// ---------------------------------------------------------------------------
__global__ __launch_bounds__(128)
void mha_flash_wmma_kernel(const float* __restrict__ pre_q,
                           const float* __restrict__ pre_v,
                           const float* __restrict__ pre_k,
                           _Float16*    __restrict__ merged) {
    __shared__ _Float16     Kl[2][32][KPAD];  // [key][dh]
    __shared__ _Float16     Vt[2][64][VPAD];  // [dh][pos] (transposed, interleaved keys)
    __shared__ unsigned int Plds[4][16][16];  // per-wave P tile, packed pairs

    const int tid  = threadIdx.x;
    const int lane = tid & 31;
    const int wid  = tid >> 5;
    const int hi   = lane >> 4;
    const int lx   = lane & 15;

    const int bh = blockIdx.x >> 5;        // 32 blocks per (b,h)
    const int qb = blockIdx.x & 31;        // q-block: rows [qb*64, qb*64+64)
    const int h  = bh & 15;
    const int b  = bh >> 4;
    const int q0 = (qb * 4 + wid) * 16;    // this wave's q tile
    const int my_jmax = q0 + 16;
    const int T = 2 * qb + 2;              // 32-key tiles needed by the block

    // ---- Q fragments (A-layout, loop-invariant) ----
    const float* qrow = pre_q + qkv_idx(b, q0 + lx, h, 0);
    v16h aq0, aq1;
#pragma unroll
    for (int i = 0; i < 8; ++i) {
        aq0[i]     = (_Float16)qrow[16 * hi + i];
        aq0[8 + i] = (_Float16)qrow[16 * hi + 8 + i];
        aq1[i]     = (_Float16)qrow[32 + 16 * hi + i];
        aq1[8 + i] = (_Float16)qrow[32 + 16 * hi + 8 + i];
    }

    v16h ones;
#pragma unroll
    for (int i = 0; i < 16; ++i) ones[i] = (_Float16)1.0f;

    v8f acc0 = {}, acc1 = {}, acc2 = {}, acc3 = {};
    float rowM[8], rowL[8];
#pragma unroll
    for (int r = 0; r < 8; ++r) { rowM[r] = -1e30f; rowL[r] = 0.0f; }

    float kreg[16], vreg[16];

    // issue coalesced global loads of tile t into registers (no LDS yet)
    auto issue_loads = [&](int t) {
#pragma unroll
        for (int it = 0; it < 16; ++it) {
            const int e   = it * 128 + tid;
            const int key = e >> 6;
            const int dh  = e & 63;
            kreg[it] = pre_k[qkv_idx(b, t * 32 + key, h, dh)];
            vreg[it] = pre_v[qkv_idx(b, t * 32 + key, h, dh)];
        }
    };
    // convert + store staged registers into LDS buffer
    auto store_stage = [&](int buf) {
#pragma unroll
        for (int it = 0; it < 16; ++it) {
            const int e   = it * 128 + tid;
            const int key = e >> 6;
            const int dh  = e & 63;
            const int pos = 2 * (key & 15) + (key >> 4);   // interleaved key order
            Kl[buf][key][dh]  = (_Float16)kreg[it];
            Vt[buf][dh][pos]  = (_Float16)vreg[it];
        }
    };

    issue_loads(0);
    for (int t = 0; t < T; ++t) {
        store_stage(t & 1);
        __syncthreads();                           // tile t visible to block
        if (t + 2 < T) {                           // speculative prefetch t+2
            __builtin_prefetch(pre_k + qkv_idx(b, (t + 2) * 32 + (tid >> 6), h, tid & 63), 0, 0);
            __builtin_prefetch(pre_v + qkv_idx(b, (t + 2) * 32 + (tid >> 6), h, tid & 63), 0, 0);
        }
        if (t + 1 < T) issue_loads(t + 1);         // overlap global latency with WMMAs

        const int j0 = t * 32;
        if (j0 < my_jmax) {
            const int buf = t & 1;

            // ---- S = Q K^T : 4 WMMAs ----
            v8f s0 = {}, s1 = {};
            s0 = wmma_f16(aq0, ld_frag_lds(&Kl[buf][lx][16 * hi]), s0);
            s1 = wmma_f16(aq0, ld_frag_lds(&Kl[buf][16 + lx][16 * hi]), s1);
            s0 = wmma_f16(aq1, ld_frag_lds(&Kl[buf][lx][32 + 16 * hi]), s0);
            s1 = wmma_f16(aq1, ld_frag_lds(&Kl[buf][16 + lx][32 + 16 * hi]), s1);

            // ---- scale + causal mask + online softmax (max only via lanes) ----
#pragma unroll
            for (int r = 0; r < 8; ++r) {
                const int qi = q0 + r + 8 * hi;
                float e0 = s0[r] * 0.125f;
                float e1 = s1[r] * 0.125f;
                e0 = ((j0 + lx)      <= qi) ? e0 : NEG_INF_F;
                e1 = ((j0 + 16 + lx) <= qi) ? e1 : NEG_INF_F;

                float mx = fmaxf(e0, e1);
#pragma unroll
                for (int off = 8; off > 0; off >>= 1)
                    mx = fmaxf(mx, __shfl_xor(mx, off, 32));

                const float mnew  = fmaxf(rowM[r], mx);
                const float alpha = __expf(rowM[r] - mnew);
                const float p0 = __expf(e0 - mnew);
                const float p1 = __expf(e1 - mnew);
                rowL[r] *= alpha;
                rowM[r]  = mnew;
                acc0[r] *= alpha; acc1[r] *= alpha; acc2[r] *= alpha; acc3[r] *= alpha;

                union { _Float16 hv[2]; unsigned int u; } pk;
                pk.hv[0] = (_Float16)p0;           // pos 2*lx   (key lx)
                pk.hv[1] = (_Float16)p1;           // pos 2*lx+1 (key 16+lx)
                Plds[wid][r + 8 * hi][lx] = pk.u;  // single ds_store_b32
            }

            // same-wave LDS RAW (C-layout stores -> A-layout loads)
            asm volatile("s_wait_dscnt 0" ::: "memory");

            const _Float16* prow = (const _Float16*)&Plds[wid][lx][0];
            v16h pa;
            {
                v8h plo = *(const v8h*)(prow + 16 * hi);
                v8h phi = *(const v8h*)(prow + 16 * hi + 8);
#pragma unroll
                for (int i = 0; i < 8; ++i) { pa[i] = plo[i]; pa[8 + i] = phi[i]; }
            }

            // ---- row sums via one WMMA: (P x ones) broadcasts sum per row ----
            v8f zero = {};
            v8f sums = wmma_f16(pa, ones, zero);
#pragma unroll
            for (int r = 0; r < 8; ++r) rowL[r] += sums[r];

            // ---- O += P V : 4 WMMAs (V^T rows are contiguous b128s) ----
            acc0 = wmma_f16(pa, ld_frag_lds(&Vt[buf][ 0 + lx][16 * hi]), acc0);
            acc1 = wmma_f16(pa, ld_frag_lds(&Vt[buf][16 + lx][16 * hi]), acc1);
            acc2 = wmma_f16(pa, ld_frag_lds(&Vt[buf][32 + lx][16 * hi]), acc2);
            acc3 = wmma_f16(pa, ld_frag_lds(&Vt[buf][48 + lx][16 * hi]), acc3);
        }
    }

    // ---- finalize: O /= l, write merged heads (f16) ----
#pragma unroll
    for (int r = 0; r < 8; ++r) {
        const float inv = 1.0f / rowL[r];
        const size_t obase = qkv_idx(b, q0 + r + 8 * hi, h, 0);
        merged[obase +  0 + lx] = (_Float16)(acc0[r] * inv);
        merged[obase + 16 + lx] = (_Float16)(acc1[r] * inv);
        merged[obase + 32 + lx] = (_Float16)(acc2[r] * inv);
        merged[obase + 48 + lx] = (_Float16)(acc3[r] * inv);
    }
}

// ---------------------------------------------------------------------------
// Kernel 2: convert output_weights f32 -> f16, transposed: WT[n][k] = W[k][n]
// ---------------------------------------------------------------------------
__global__ __launch_bounds__(256)
void wt_cvt_kernel(const float* __restrict__ W, _Float16* __restrict__ WT) {
    const int id = blockIdx.x * 256 + threadIdx.x;   // 0 .. 1M-1
    const int k  = id >> 10;
    const int n  = id & 1023;
    WT[(size_t)n * DM + k] = (_Float16)W[id];
}

// ---------------------------------------------------------------------------
// Kernel 3: out(4096x1024 f32) = merged(f16) x W.
// Block = 256 threads (8 waves) -> 128x64 output tile. The 64x32 WT tile for
// each K chunk is DMA'd into LDS by the Tensor Data Mover (wave 0 issues,
// double-buffered via TENSORcnt), with TDM padding giving 40-half LDS rows.
// ---------------------------------------------------------------------------
__global__ __launch_bounds__(256)
void out_proj_wmma_kernel(const _Float16* __restrict__ A,
                          const _Float16* __restrict__ WT,
                          float*          __restrict__ out) {
    __shared__ _Float16 Wl[2][64][VPAD];   // [col][k], TDM-padded rows

    const int tid  = threadIdx.x;
    const int lane = tid & 31;
    const int wid  = tid >> 5;
    const int hi   = lane >> 4;
    const int lx   = lane & 15;

    const int rowBlk = blockIdx.x >> 4;    // 32 row blocks of 128
    const int colGrp = blockIdx.x & 15;    // 16 col groups of 64
    const int row0 = rowBlk * 128 + wid * 16;
    const int col0 = colGrp * 64;

    v8f acc[4] = {};

    // TDM: load WT tile [col0..col0+63] x [k0..k0+31] (f16) into Wl[buf]
    auto tdm_issue = [&](int kc, int buf) {
        const unsigned long long ga =
            (unsigned long long)(uintptr_t)(WT + (size_t)col0 * DM + (size_t)kc * 32);
        const unsigned int lds = (unsigned int)(uintptr_t)&Wl[buf][0][0];
        v4u g0;
        g0[0] = 1u;                                   // count=1 (valid user D#)
        g0[1] = lds;                                  // lds_addr
        g0[2] = (unsigned int)(ga & 0xffffffffu);     // global_addr[31:0]
        g0[3] = (unsigned int)((ga >> 32) & 0x01ffffffu) | (2u << 30); // addr[56:32], type=2
        v8i g1;
        g1[0] = (int)((1u << 16)        // data_size = 2 bytes
                    | (1u << 20)        // pad_enable
                    | (3u << 22)        // pad_interval: 16 DWORDs (= one 32-half row)
                    | (3u << 25));      // pad_amount: 4 DWORDs (= 8 halves -> 40/row)
        g1[1] = (int)(1024u << 16);     // tensor_dim0 = 1024 (k extent)
        g1[2] = (int)(1024u << 16);     // tensor_dim0 hi=0 | tensor_dim1 = 1024 (n extent)
        g1[3] = (int)(32u << 16);       // tensor_dim1 hi=0 | tile_dim0 = 32 (k chunk)
        g1[4] = 64;                     // tile_dim1 = 64 (cols) | tile_dim2 = 0
        g1[5] = 1024;                   // tensor_dim0_stride = 1024
        g1[6] = 0;
        g1[7] = 0;
        v4i gz = {0, 0, 0, 0};          // 2-D tile: groups 2/3 unused
        v8i gz8 = {0, 0, 0, 0, 0, 0, 0, 0};
        // amdgpu-toolchain (clang-23) 6-arg form: (g0, g1, g2, g3, g4, cpol)
        __builtin_amdgcn_tensor_load_to_lds(g0, g1, gz, gz, gz8, 0);
    };

    if (wid == 0) tdm_issue(0, 0);
    for (int kc = 0; kc < 32; ++kc) {
        if (wid == 0) {
            if (kc + 1 < 32) {
                tdm_issue(kc + 1, (kc + 1) & 1);
                __builtin_amdgcn_s_wait_tensorcnt(1);   // tile kc landed (in-order)
            } else {
                __builtin_amdgcn_s_wait_tensorcnt(0);
            }
        }
        __syncthreads();                                // tile kc visible to block
        const int buf = kc & 1;

        // A fragment: two global b128 loads from f16 workspace
        const _Float16* arow = A + (size_t)(row0 + lx) * DM + 32 * kc + 16 * hi;
        v16h a;
        {
            v8h alo = *(const v8h*)arow;
            v8h ahi = *(const v8h*)(arow + 8);
#pragma unroll
            for (int i = 0; i < 8; ++i) { a[i] = alo[i]; a[8 + i] = ahi[i]; }
        }
        if (kc + 1 < 32)
            __builtin_prefetch(arow + 32, 0, 0);

#pragma unroll
        for (int t = 0; t < 4; ++t)
            acc[t] = wmma_f16(a, ld_frag_lds(&Wl[buf][16 * t + lx][16 * hi]), acc[t]);

        __syncthreads();                                // block done with buf before TDM reuses it
    }

#pragma unroll
    for (int t = 0; t < 4; ++t) {
#pragma unroll
        for (int r = 0; r < 8; ++r) {
            out[(size_t)(row0 + r + 8 * hi) * DM + col0 + 16 * t + lx] = acc[t][r];
        }
    }
}

extern "C" void kernel_launch(void* const* d_in, const int* in_sizes, int n_in,
                              void* d_out, int out_size, void* d_ws, size_t ws_size,
                              hipStream_t stream) {
    (void)in_sizes; (void)n_in; (void)out_size; (void)ws_size;
    const float* pre_q = (const float*)d_in[0];
    const float* pre_v = (const float*)d_in[1];
    const float* pre_k = (const float*)d_in[2];
    const float* ow    = (const float*)d_in[3];
    float*       out   = (float*)d_out;

    _Float16* merged = (_Float16*)d_ws;                      // 8 MB
    _Float16* wt     = merged + (size_t)BB * SS * DM;        // +2 MB

    wt_cvt_kernel<<<(DM * DM) / 256, 256, 0, stream>>>(ow, wt);
    mha_flash_wmma_kernel<<<1024, 128, 0, stream>>>(pre_q, pre_v, pre_k, merged);
    out_proj_wmma_kernel<<<512, 256, 0, stream>>>(merged, wt, out);
}
